// ResnetBlocWithCrossAttn_55130200211991
// MI455X (gfx1250) — matrix operand order repaired and checked by reading, not compiled, not run on hardware
//
#include <hip/hip_runtime.h>
#include <hip/hip_bf16.h>

typedef __attribute__((ext_vector_type(16))) _Float16 v16h;
typedef __attribute__((ext_vector_type(8)))  float    v8f;

#define BATCH 8
#define CCH   192
#define HH    48
#define WW    48
#define HWP   (HH*WW)            // 2304
#define MTOT  (BATCH*HWP)        // 18432
#define GROUPS 32
#define CPG    (CCH/GROUPS)      // 6
#define PH    51                 // padded rows (halo + out-conv K-pad overrun row)
#define PW    50                 // padded cols

union FragH { v16h v; unsigned u[8]; _Float16 h[16]; };
union Acc8  { v8f  v; float f[8]; };

// single-instruction max (skips fmaxf NaN-canonicalize churn)
__device__ __forceinline__ float vmaxf(float a, float b) {
    float d;
    asm("v_max_num_f32 %0, %1, %2" : "=v"(d) : "v"(a), "v"(b));
    return d;
}

// ---------------------------------------------------------------------------
// v_permlane16_b32 XOR-butterfly row-max: reduce across each 16-lane group
// (pure VALU crossbar -- no LDS, no dscnt waits).  Selector nibbles encode
// src_lane = lane ^ m for lanes 0-7 (lo) and 8-15 (hi); same for both halves.
// ---------------------------------------------------------------------------
__device__ __forceinline__ float plane16(float x, unsigned lo, unsigned hi) {
    union { float f; unsigned u; } a, r;
    a.f = x;
    r.u = __builtin_amdgcn_permlane16(a.u, a.u, lo, hi, false, false);
    return r.f;
}
__device__ __forceinline__ float red16_max(float x) {
    x = vmaxf(x, plane16(x, 0x67452301u, 0xEFCDAB89u));  // ^1
    x = vmaxf(x, plane16(x, 0x54761032u, 0xDCFE98BAu));  // ^2
    x = vmaxf(x, plane16(x, 0x32107654u, 0xBA98FEDCu));  // ^4
    x = vmaxf(x, plane16(x, 0xFEDCBA98u, 0x76543210u));  // ^8
    return x;
}

// ---------------------------------------------------------------------------
// zero-fill (u32 granularity) for padded activation buffers
// ---------------------------------------------------------------------------
__global__ void zfill_k(unsigned* __restrict__ p, int n) {
    int i = blockIdx.x * blockDim.x + threadIdx.x;
    if (i < n) p[i] = 0u;
}

// ---------------------------------------------------------------------------
// Weight repack: OIHW f32 -> [co][KPAD] f16 with k = (r*3+s)*CIN + ci, zero pad
// ---------------------------------------------------------------------------
template<int CIN>
__global__ void wcvt_k(const float* __restrict__ w, _Float16* __restrict__ wt) {
    constexpr int KTOT = CIN * 9;
    constexpr int KPAD = (KTOT + 31) & ~31;
    int idx = blockIdx.x * blockDim.x + threadIdx.x;
    if (idx >= CCH * KPAD) return;
    int co = idx / KPAD, k = idx % KPAD;
    float v = 0.f;
    if (k < KTOT) {
        int rs = k / CIN, ci = k - rs * CIN;
        int r = rs / 3, s = rs - 3 * r;
        v = w[((co * CIN + ci) * 3 + r) * 3 + s];
    }
    wt[idx] = (_Float16)v;
}

// ---------------------------------------------------------------------------
// FeatureWiseAffine: gbsum[b,c] = 0.5*((emb@aff_w.T+aff_b)[c] + (...)[c+192])
// ---------------------------------------------------------------------------
__global__ void affine_k(const float* __restrict__ emb, const float* __restrict__ aw,
                         const float* __restrict__ ab, float* __restrict__ gbs) {
    int idx = blockIdx.x * blockDim.x + threadIdx.x;
    if (idx >= BATCH * CCH) return;
    int b = idx / CCH, c = idx % CCH;
    float s1 = ab[c], s2 = ab[c + CCH];
    for (int i = 0; i < CCH; ++i) {
        float e = emb[b * CCH + i];
        s1 += e * aw[c * CCH + i];
        s2 += e * aw[(c + CCH) * CCH + i];
    }
    gbs[idx] = 0.5f * (s1 + s2);
}

// ---------------------------------------------------------------------------
// GroupNorm statistics: per (b,g) mean + rstd.  layout 0=NCHW, 1=NHWC
// ---------------------------------------------------------------------------
__global__ void gn_stats_k(const float* __restrict__ x, float* __restrict__ mr, int layout) {
    __shared__ float r1[256], r2[256];
    int b = blockIdx.x / GROUPS, g = blockIdx.x % GROUPS;
    float s = 0.f, ss = 0.f;
    for (int i = threadIdx.x; i < CPG * HWP; i += 256) {
        int c = g * CPG + i / HWP, p = i % HWP;
        float v = (layout == 0) ? x[(b * CCH + c) * HWP + p]
                                : x[(b * HWP + p) * CCH + c];
        s += v; ss += v * v;
    }
    r1[threadIdx.x] = s; r2[threadIdx.x] = ss;
    __syncthreads();
    for (int off = 128; off > 0; off >>= 1) {
        if (threadIdx.x < off) { r1[threadIdx.x] += r1[threadIdx.x + off];
                                 r2[threadIdx.x] += r2[threadIdx.x + off]; }
        __syncthreads();
    }
    if (threadIdx.x == 0) {
        float n = (float)(CPG * HWP);
        float mean = r1[0] / n;
        float var  = r2[0] / n - mean * mean;
        mr[blockIdx.x * 2]     = mean;
        mr[blockIdx.x * 2 + 1] = rsqrtf(var + 1e-5f);
    }
}

// ---------------------------------------------------------------------------
// GroupNorm apply + swish -> f16, written into halo-padded NHWC [B][51][50][C]
// input layout: 0 = NCHW f32, 1 = NHWC (unpadded) f32
// ---------------------------------------------------------------------------
__global__ void gn_apply_k(const float* __restrict__ x, const float* __restrict__ mr,
                           const float* __restrict__ gw, const float* __restrict__ gb,
                           _Float16* __restrict__ outp, int layout) {
    int idx = blockIdx.x * blockDim.x + threadIdx.x;  // NHWC linear (unpadded)
    if (idx >= MTOT * CCH) return;
    int c = idx % CCH;
    int p = (idx / CCH) % HWP;
    int b = idx / (CCH * HWP);
    float v = (layout == 0) ? x[(b * CCH + c) * HWP + p] : x[idx];
    int g = c / CPG;
    float m = mr[(b * GROUPS + g) * 2], r = mr[(b * GROUPS + g) * 2 + 1];
    v = (v - m) * r * gw[c] + gb[c];
    int h = p / WW, w = p % WW;
    outp[((size_t)((b * PH + h + 1) * PW) + (w + 1)) * CCH + c] =
        (_Float16)(v / (1.f + __expf(-v)));
}

// ---------------------------------------------------------------------------
// Implicit-GEMM 3x3 conv (pad=1): halo-padded NHWC f16 act x repacked f16
// weights.  One wave = 16(M-pixels) x 32(N-cout) tile; fully unrolled K loop,
// all source offsets constant-folded so the inner loop is pure b128 loads +
// v_wmma_f32_16x16x32_f16 (A reused across 2 N-subtiles).
// mode: 0 = conv1 (+bias, FW-affine, NHWC f32 out)
//       1 = conv2 (+bias + x(NCHW), NHWC f32 out = xres)
//       2 = out-conv (+bias + xres(NHWC), NCHW f32 out)
// ---------------------------------------------------------------------------
template<int CIN>
__global__ __launch_bounds__(256) void conv3_gemm_k(
    const _Float16* __restrict__ act, const _Float16* __restrict__ wt,
    const float* __restrict__ bias, const float* __restrict__ gbs,
    const float* __restrict__ res, float* __restrict__ outp, int mode) {
    constexpr int KTOT = CIN * 9;
    constexpr int KPAD = (KTOT + 31) & ~31;
    int lane = threadIdx.x & 31, wid = threadIdx.x >> 5;
    int half = lane >> 4, l16 = lane & 15;
    int tile = blockIdx.x * 8 + wid;                 // grid is exact
    int nt = tile % (CCH / 32), mtile = tile / (CCH / 32);
    int col = nt * 32 + l16;
    int pixA = mtile * 16 + l16;
    int b = pixA / HWP;                              // 2304 % 16 == 0: uniform
    int hwA = pixA % HWP;
    int h = hwA / WW, w = hwA % WW;
    // per-lane A base at padded (h, w) == unpadded (h-1, w-1); lane half-term folded in
    const _Float16* ap  = act + ((size_t)((b * PH + h) * PW) + w) * CIN + half * 8;
    const _Float16* bp0 = wt + (size_t)col * KPAD + half * 8;
    const _Float16* bp1 = bp0 + (size_t)16 * KPAD;
    __builtin_prefetch(bp0, 0, 1);
    __builtin_prefetch(bp1, 0, 1);
    Acc8 acc0 = {}, acc1 = {};
#pragma unroll
    for (int kb = 0; kb < KPAD; kb += 32) {
        FragH a, b0, b1;
#pragma unroll
        for (int j = 0; j < 8; ++j) {
            // A 16x32 f16 layout: k = kb + (j<4 ? half*8+2j : 16+half*8+2(j-4));
            // half*8 lives in ap, the rest is a compile-time constant:
            int kpos = kb + ((j < 4) ? 0 : 16) + 2 * (j & 3);
            int rs = kpos / CIN, cic = kpos % CIN;   // folds to constants
            int r = rs / 3, sxy = rs - 3 * r;
            a.u[j]  = *(const unsigned*)(ap + ((r * PW + sxy) * CIN + cic));
            // B 32x16 f16 layout: same k as A (half*8 folded into bp)
            b0.u[j] = *(const unsigned*)(bp0 + (kb + ((j < 4) ? 0 : 16) + 2 * (j & 3)));
            b1.u[j] = *(const unsigned*)(bp1 + (kb + ((j < 4) ? 0 : 16) + 2 * (j & 3)));
        }
        acc0.v = __builtin_amdgcn_wmma_f32_16x16x32_f16(
            false, a.v, false, b0.v, (short)0, acc0.v, false, false);
        acc1.v = __builtin_amdgcn_wmma_f32_16x16x32_f16(
            false, a.v, false, b1.v, (short)0, acc1.v, false, false);
    }
    float bs0 = bias[col], bs1 = bias[col + 16];
#pragma unroll
    for (int j = 0; j < 8; ++j) {
        int pix = mtile * 16 + half * 8 + j;         // C layout: VGPR j -> row 8*half+j
        int hw = pix % HWP;
        float v0 = acc0.f[j] + bs0;
        float v1 = acc1.f[j] + bs1;
        if (mode == 0) {
            float s0 = gbs[b * CCH + col], s1 = gbs[b * CCH + col + 16];
            v0 = v0 * (1.f + s0) + s0;
            v1 = v1 * (1.f + s1) + s1;
            outp[(size_t)pix * CCH + col]      = v0;
            outp[(size_t)pix * CCH + col + 16] = v1;
        } else if (mode == 1) {
            v0 += res[((size_t)b * CCH + col) * HWP + hw];
            v1 += res[((size_t)b * CCH + col + 16) * HWP + hw];
            outp[(size_t)pix * CCH + col]      = v0;
            outp[(size_t)pix * CCH + col + 16] = v1;
        } else {
            v0 += res[(size_t)pix * CCH + col];
            v1 += res[(size_t)pix * CCH + col + 16];
            outp[((size_t)b * CCH + col) * HWP + hw]      = v0;
            outp[((size_t)b * CCH + col + 16) * HWP + hw] = v1;
        }
    }
}

// ---------------------------------------------------------------------------
// 1x1 projections (bandwidth-trivial)
// ---------------------------------------------------------------------------
__global__ void q_proj_k(const float* __restrict__ xres, const float* __restrict__ qw,
                         _Float16* __restrict__ Q) {
    int idx = blockIdx.x * blockDim.x + threadIdx.x;
    if (idx >= BATCH * HWP * 16) return;
    int c = idx % 16, p = (idx / 16) % HWP, b = idx / (16 * HWP);
    float s = 0.f;
    for (int ci = 0; ci < CCH; ++ci)
        s += xres[((size_t)b * HWP + p) * CCH + ci] * qw[c * CCH + ci];
    Q[idx] = (_Float16)(0.25f * s);   // scale 1/sqrt(16) folded into Q
}

// K stored [B][YX][16] (d contiguous); V stored TRANSPOSED [B][16][YX]
// (yx contiguous) so attention B-fragment (k,k+1) pairs are single u32 loads.
__global__ void kv_proj_k(const float* __restrict__ xe, const float* __restrict__ kvw,
                          _Float16* __restrict__ K, _Float16* __restrict__ Vt) {
    int idx = blockIdx.x * blockDim.x + threadIdx.x;
    if (idx >= BATCH * HWP * 16) return;
    int c = idx % 16, p = (idx / 16) % HWP, b = idx / (16 * HWP);
    float sk = 0.f, sv = 0.f;
    for (int ci = 0; ci < CCH; ++ci) {
        float e = xe[((size_t)b * CCH + ci) * HWP + p];
        sk += e * kvw[c * CCH + ci];
        sv += e * kvw[(c + 16) * CCH + ci];
    }
    K[idx] = (_Float16)sk;
    Vt[((size_t)b * 16 + c) * HWP + p] = (_Float16)sv;
}

// ---------------------------------------------------------------------------
// Flash cross-attention: one wave per 16-row Q block, 144 column tiles of 16.
// S = Q*K^T in one WMMA (d=16 zero-padded to K=32, scale folded into Q),
// online softmax: row max via v_permlane16 butterflies (pure VALU); row SUMS
// via an extra WMMA of the A-layout P fragment against an all-ones B matrix
// (C layout replicates each row sum across lanes -- exactly the shape the
// l-update needs).  P bounced through 512B/wave LDS (wave-private: single
// s_wait_dscnt, no block barrier) to convert C-layout -> A-layout, then
// O += P*V via WMMA.  Output into halo-padded NHWC [B][51][50][16].
// ---------------------------------------------------------------------------
__global__ __launch_bounds__(256) void attn_k(
    const _Float16* __restrict__ Q, const _Float16* __restrict__ K,
    const _Float16* __restrict__ Vt, _Float16* __restrict__ O) {
    __shared__ _Float16 psh[8][256];
    int lane = threadIdx.x & 31, wid = threadIdx.x >> 5;
    int half = lane >> 4, l16 = lane & 15;
    int rb = blockIdx.x * 8 + wid;            // 0..1151, grid exact
    int b = rb / (HWP / 16), mt = rb % (HWP / 16);
    int row0 = mt * 16;
    FragH qa;
#pragma unroll
    for (int j = 0; j < 8; ++j) {
        // k = half*16 + 2j; only half 0 (k<16) is real head_dim
        unsigned t = *(const unsigned*)(Q + (((size_t)b * HWP + row0 + l16) * 16 + 2 * j));
        qa.u[j] = (half == 0) ? t : 0u;       // K=16..31 zero pad
    }
    FragH onesb;                               // B matrix of 1.0h (for row sums)
#pragma unroll
    for (int j = 0; j < 8; ++j) onesb.u[j] = 0x3C003C00u;
    Acc8 o = {};
    float rm[8], rl[8];
#pragma unroll
    for (int j = 0; j < 8; ++j) { rm[j] = -1e30f; rl[j] = 0.f; }

    const _Float16* kb_base = K  + (size_t)b * HWP * 16;
    const _Float16* vt_base = Vt + ((size_t)b * 16 + l16) * HWP;

    for (int jt = 0; jt < HWP / 16; ++jt) {
        int c0 = jt * 16;
        FragH kf;
#pragma unroll
        for (int j = 0; j < 8; ++j) {
            // B layout k = half*16 + 2j; only k<16 is real head_dim.
            // Unconditional in-bounds load from (k & 15) == 2j, then zero the
            // upper half with a per-lane select (v_cndmask, no exec branch).
            unsigned t = *(const unsigned*)(kb_base + ((size_t)(c0 + l16) * 16 + 2 * j));
            kf.u[j] = (half == 0) ? t : 0u;
        }
        Acc8 z = {};
        Acc8 s;
        s.v = __builtin_amdgcn_wmma_f32_16x16x32_f16(
            false, qa.v, false, kf.v, (short)0, z.v, false, false);
#pragma unroll
        for (int j = 0; j < 8; ++j) {
            int m = half * 8 + j;
            float mx    = red16_max(s.f[j]);
            float mnew  = vmaxf(rm[j], mx);
            float p     = __expf(s.f[j] - mnew);
            float alpha = __expf(rm[j] - mnew);
            rl[j] *= alpha;                    // row sum added below via WMMA
            rm[j]  = mnew;
            o.f[j] *= alpha;
            psh[wid][m * 16 + l16] = (_Float16)p;
        }
        // psh is wave-private; LDS ops from one wave are in-order, so only the
        // store->read visibility needs a wait (no block barrier).
        asm volatile("s_wait_dscnt 0x0" ::: "memory");
        FragH pa, vf;
#pragma unroll
        for (int j = 0; j < 8; ++j) {
            {                                 // A-layout reload of P from LDS (k = half*16 + 2j)
                unsigned tp = *(const unsigned*)(&psh[wid][l16 * 16 + 2 * j]);
                pa.u[j] = (half == 0) ? tp : 0u;
            }
            // B layout: V[c0+k][n] = Vt[n][c0+k]; pair (k,k+1) contiguous in Vt.
            unsigned t = *(const unsigned*)(vt_base + (c0 + 2 * j));
            vf.u[j] = (half == 0) ? t : 0u;
        }
        // Row sums of P in one WMMA: C[m,n] = sum_k P[m,k] * 1
        Acc8 rs_;
        Acc8 z2 = {};
        rs_.v = __builtin_amdgcn_wmma_f32_16x16x32_f16(
            false, pa.v, false, onesb.v, (short)0, z2.v, false, false);
#pragma unroll
        for (int j = 0; j < 8; ++j) rl[j] += rs_.f[j];
        o.v = __builtin_amdgcn_wmma_f32_16x16x32_f16(
            false, pa.v, false, vf.v, (short)0, o.v, false, false);
    }
#pragma unroll
    for (int j = 0; j < 8; ++j) {
        int row = row0 + half * 8 + j;        // hw index within batch b
        int h = row / WW, w = row % WW;
        O[((size_t)((b * PH + h + 1) * PW) + (w + 1)) * 16 + l16] =
            (_Float16)(o.f[j] / rl[j]);
    }
}

// ---------------------------------------------------------------------------
extern "C" void kernel_launch(void* const* d_in, const int* in_sizes, int n_in,
                              void* d_out, int out_size, void* d_ws, size_t ws_size,
                              hipStream_t stream) {
    const float* x       = (const float*)d_in[0];
    const float* xe      = (const float*)d_in[1];
    const float* temb    = (const float*)d_in[2];
    const float* gn1_w   = (const float*)d_in[3];
    const float* gn1_b   = (const float*)d_in[4];
    const float* conv1_w = (const float*)d_in[5];
    const float* conv1_b = (const float*)d_in[6];
    const float* aff_w   = (const float*)d_in[7];
    const float* aff_b   = (const float*)d_in[8];
    const float* gn2_w   = (const float*)d_in[9];
    const float* gn2_b   = (const float*)d_in[10];
    const float* conv2_w = (const float*)d_in[11];
    const float* conv2_b = (const float*)d_in[12];
    const float* q_w     = (const float*)d_in[13];
    const float* kv_w    = (const float*)d_in[14];
    const float* out_w   = (const float*)d_in[15];
    const float* out_b   = (const float*)d_in[16];
    float* outp = (float*)d_out;

    // workspace carve-up (~48 MB), 256B aligned
    char* ws = (char*)d_ws;
    size_t off = 0;
    auto carve = [&](size_t bytes) { char* p = ws + off; off = (off + bytes + 255) & ~(size_t)255; return p; };
    const size_t PADN  = (size_t)BATCH * PH * PW * CCH;   // padded f16 act elems
    const size_t PADNQ = (size_t)BATCH * PH * PW * 16;    // padded f16 attn-out elems
    float*     mr1  = (float*)    carve(BATCH * GROUPS * 2 * 4);
    float*     mr2  = (float*)    carve(BATCH * GROUPS * 2 * 4);
    float*     gbs  = (float*)    carve(BATCH * CCH * 4);
    _Float16*  wt1  = (_Float16*) carve((size_t)CCH * 1728 * 2);
    _Float16*  wt2  = (_Float16*) carve((size_t)CCH * 1728 * 2);
    _Float16*  wt3  = (_Float16*) carve((size_t)CCH * 160 * 2);
    _Float16*  a1   = (_Float16*) carve(PADN * 2);
    float*     h1   = (float*)    carve((size_t)MTOT * CCH * 4);
    _Float16*  a2   = (_Float16*) carve(PADN * 2);
    float*     xres = (float*)    carve((size_t)MTOT * CCH * 4);
    _Float16*  Qb   = (_Float16*) carve((size_t)MTOT * 16 * 2);
    _Float16*  Kb   = (_Float16*) carve((size_t)MTOT * 16 * 2);
    _Float16*  Vtb  = (_Float16*) carve((size_t)MTOT * 16 * 2);
    _Float16*  AOp  = (_Float16*) carve(PADNQ * 2);
    (void)ws_size; (void)in_sizes; (void)n_in; (void)out_size;

    const int elemN  = MTOT * CCH;                  // 3,538,944
    const int tiles2 = (MTOT / 16) * (CCH / 32);    // 6,912 wave-tiles per conv

    // zero padded activation buffers (halo must be 0)
    zfill_k<<<(int)((PADN / 2 + 255) / 256), 256, 0, stream>>>((unsigned*)a1, (int)(PADN / 2));
    zfill_k<<<(int)((PADN / 2 + 255) / 256), 256, 0, stream>>>((unsigned*)a2, (int)(PADN / 2));
    zfill_k<<<(int)((PADNQ / 2 + 255) / 256), 256, 0, stream>>>((unsigned*)AOp, (int)(PADNQ / 2));
    // weight repack
    wcvt_k<192><<<(CCH * 1728 + 255) / 256, 256, 0, stream>>>(conv1_w, wt1);
    wcvt_k<192><<<(CCH * 1728 + 255) / 256, 256, 0, stream>>>(conv2_w, wt2);
    wcvt_k<16> <<<(CCH * 160  + 255) / 256, 256, 0, stream>>>(out_w,   wt3);
    // time-embedding affine
    affine_k<<<(BATCH * CCH + 255) / 256, 256, 0, stream>>>(temb, aff_w, aff_b, gbs);
    // GN1 + swish -> a1 (padded f16 NHWC)
    gn_stats_k<<<BATCH * GROUPS, 256, 0, stream>>>(x, mr1, 0);
    gn_apply_k<<<elemN / 256, 256, 0, stream>>>(x, mr1, gn1_w, gn1_b, a1, 0);
    // conv1 + bias + FeatureWiseAffine -> h1 (f32 NHWC)
    conv3_gemm_k<192><<<tiles2 / 8, 256, 0, stream>>>(a1, wt1, conv1_b, gbs, nullptr, h1, 0);
    // GN2 + swish -> a2
    gn_stats_k<<<BATCH * GROUPS, 256, 0, stream>>>(h1, mr2, 1);
    gn_apply_k<<<elemN / 256, 256, 0, stream>>>(h1, mr2, gn2_w, gn2_b, a2, 1);
    // conv2 + bias + x -> xres (f32 NHWC)
    conv3_gemm_k<192><<<tiles2 / 8, 256, 0, stream>>>(a2, wt2, conv2_b, nullptr, x, xres, 1);
    // q / kv projections (f16; V transposed)
    q_proj_k <<<(MTOT * 16) / 256, 256, 0, stream>>>(xres, q_w, Qb);
    kv_proj_k<<<(MTOT * 16) / 256, 256, 0, stream>>>(xe, kv_w, Kb, Vtb);
    // flash cross-attention -> AOp (padded f16 NHWC)
    attn_k<<<(MTOT / 16) / 8, 256, 0, stream>>>(Qb, Kb, Vtb, AOp);
    // out-conv + bias + xres -> final (f32 NCHW)
    conv3_gemm_k<16><<<tiles2 / 8, 256, 0, stream>>>(AOp, wt3, out_b, nullptr, xres, outp, 2);
}
